// RNN_Variational_Decoder_85383949844740
// MI455X (gfx1250) — compile-verified, matrix operand order checked by reading, not run on hardware
//
#include <hip/hip_runtime.h>
#include <hip/hip_bf16.h>

typedef _Float16 h16;
typedef __attribute__((ext_vector_type(16))) _Float16 v16h;
typedef __attribute__((ext_vector_type(8)))  _Float16 v8h;
typedef __attribute__((ext_vector_type(8)))  float    v8f;

#define B_   256
#define TMAX 512
#define H_   512
#define O_   128
#define M_   512
#define F_   256

#define LDP_H 520   // padded LDS stride for 512-wide f16 tiles (1040B = 65*16B)
#define LDP_X 136   // padded LDS stride for 128-wide f16 tiles (272B = 17*16B)

__device__ __forceinline__ float sigm(float x) { return 1.0f / (1.0f + expf(-x)); }

// Load a 16x32 f16 WMMA operand fragment from a row-major matrix (ld in halves).
// A operand: row = batch row. B operand: row = output neuron (column-per-lane layout).
// ISA 16-bit A layout: lane<16 -> K {0..7,16..23}; lane>=16 -> K {8..15,24..31}.
__device__ __forceinline__ v16h load_frag(const h16* base, int ld, int row0, int k0, int lane) {
    int r  = lane & 15;
    int ko = (lane >> 4) << 3;
    const h16* p = base + (size_t)(row0 + r) * ld + k0 + ko;
    union { struct { v8h lo, hi; } s; v16h v; } u;
    u.s.lo = *(const v8h*)p;
    u.s.hi = *(const v8h*)(p + 16);
    return u.v;
}

__device__ __forceinline__ v8f wmma16(v16h a, v16h b, v8f c) {
    return __builtin_amdgcn_wmma_f32_16x16x32_f16(false, a, false, b, (short)0, c, false, false);
}

// ---------------- setup kernels ----------------

__global__ void k_f32_to_f16(const float* __restrict__ s, h16* __restrict__ d, int n) {
    int i = blockIdx.x * 256 + threadIdx.x;
    if (i < n) d[i] = (h16)s[i];
}

__global__ void k_zero_x(h16* __restrict__ x) {
    int i = blockIdx.x * 256 + threadIdx.x;
    if (i < B_ * O_) x[i] = (h16)0.0f;
}

// meta[t] = exclusive prefix of batch sizes bs[t] = #(lengths > t); meta[TMAX] = N
__global__ void k_meta(const int* __restrict__ lengths, int* __restrict__ meta) {
    __shared__ int bs[TMAX];
    int t = threadIdx.x;
    int cnt = 0;
    for (int b = 0; b < B_; ++b) cnt += (lengths[b] > t) ? 1 : 0;
    bs[t] = cnt;
    __syncthreads();
    if (t == 0) {
        int acc = 0;
        for (int i = 0; i < TMAX; ++i) { meta[i] = acc; acc += bs[i]; }
        meta[TMAX] = acc;
    }
}

// h0/c0 = (features @ W_f2h^T + b).reshape(-1,H,2) split even/odd columns
__global__ void k_init(const h16* __restrict__ feat, const h16* __restrict__ Wf2h,
                       const float* __restrict__ bf2h,
                       h16* __restrict__ hbuf, float* __restrict__ cbuf) {
    int lane = threadIdx.x & 31, wv = threadIdx.x >> 5;
    int b0 = blockIdx.x * 16;
    int c0 = blockIdx.y * 512 + wv * 64;
    v8f acc[4] = {};
    for (int kc = 0; kc < F_; kc += 32) {
        v16h a = load_frag(feat, F_, b0, kc, lane);
        #pragma unroll
        for (int nt = 0; nt < 4; ++nt) {
            v16h bf = load_frag(Wf2h, F_, c0 + nt * 16, kc, lane);
            acc[nt] = wmma16(a, bf, acc[nt]);
        }
    }
    int rowoff = (lane >> 4) * 8, col_l = lane & 15;
    #pragma unroll
    for (int nt = 0; nt < 4; ++nt) {
        int n = c0 + nt * 16 + col_l;
        float bb = bf2h[n];
        int j = n >> 1;
        #pragma unroll
        for (int r = 0; r < 8; ++r) {
            int b = b0 + rowoff + r;
            float v = acc[nt][r] + bb;
            if (n & 1) cbuf[b * H_ + j] = v;
            else       hbuf[b * H_ + j] = (h16)v;
        }
    }
}

// ---------------- per-timestep kernels ----------------

// gates = x@W_ih^T + h@W_hh^T (+biases in epilogue), then LSTM cell update.
// Workgroup owns (16 batch rows) x (64 hidden cols) across all 4 gate sections.
// A-operands (x tile, h tile) are staged in LDS once per block; WMMA A-fragments
// then come from ds_load_b128 instead of 8x-redundant global traffic.
__global__ void k_step_gates(const h16* __restrict__ xin, const h16* __restrict__ hprev,
                             h16* __restrict__ hnext, float* __restrict__ cbuf,
                             const h16* __restrict__ Wih, const h16* __restrict__ Whh,
                             const float* __restrict__ bih, const float* __restrict__ bhh) {
    __shared__ alignas(16) h16 h_s[16][LDP_H];
    __shared__ alignas(16) h16 x_s[16][LDP_X];
    __shared__ float gl[16][256];
    int lane = threadIdx.x & 31, wv = threadIdx.x >> 5;
    int b0 = blockIdx.x * 16;
    int j0 = blockIdx.y * 64;

    // stage A tiles: h (16x512 halves, 64 chunks/row) and x (16x128 halves, 16 chunks/row)
    for (int i = threadIdx.x; i < 16 * 64; i += 256) {
        int row = i >> 6;
        int c8  = (i & 63) << 3;
        *(v8h*)&h_s[row][c8] = *(const v8h*)&hprev[(size_t)(b0 + row) * H_ + c8];
    }
    for (int i = threadIdx.x; i < 16 * 16; i += 256) {
        int row = i >> 4;
        int c8  = (i & 15) << 3;
        *(v8h*)&x_s[row][c8] = *(const v8h*)&xin[(size_t)(b0 + row) * O_ + c8];
    }
    __syncthreads();

    int s   = wv >> 1;          // gate section i/f/g/o
    int sub = (wv & 1) * 32;    // column half within section
    int nrow0 = s * H_ + j0 + sub;
    v8f acc[2] = {};
    for (int kc = 0; kc < O_; kc += 32) {
        v16h a = load_frag(&x_s[0][0], LDP_X, 0, kc, lane);
        #pragma unroll
        for (int nt = 0; nt < 2; ++nt) {
            v16h bf = load_frag(Wih, O_, nrow0 + nt * 16, kc, lane);
            acc[nt] = wmma16(a, bf, acc[nt]);
        }
    }
    for (int kc = 0; kc < H_; kc += 32) {
        v16h a = load_frag(&h_s[0][0], LDP_H, 0, kc, lane);
        #pragma unroll
        for (int nt = 0; nt < 2; ++nt) {
            v16h bf = load_frag(Whh, H_, nrow0 + nt * 16, kc, lane);
            acc[nt] = wmma16(a, bf, acc[nt]);
        }
    }
    int rowoff = (lane >> 4) * 8, col_l = lane & 15;
    #pragma unroll
    for (int nt = 0; nt < 2; ++nt) {
        int lcol = s * 64 + sub + nt * 16 + col_l;
        #pragma unroll
        for (int r = 0; r < 8; ++r) gl[rowoff + r][lcol] = acc[nt][r];
    }
    __syncthreads();
    for (int q = 0; q < 4; ++q) {
        int e  = threadIdx.x + 256 * q;
        int r  = e >> 6;
        int lj = e & 63;
        int b = b0 + r, jg = j0 + lj;
        float iv = gl[r][lj]       + bih[jg]          + bhh[jg];
        float fv = gl[r][64 + lj]  + bih[H_ + jg]     + bhh[H_ + jg];
        float gv = gl[r][128 + lj] + bih[2 * H_ + jg] + bhh[2 * H_ + jg];
        float ov = gl[r][192 + lj] + bih[3 * H_ + jg] + bhh[3 * H_ + jg];
        float co = cbuf[b * H_ + jg];
        float cn = sigm(fv) * co + sigm(iv) * tanhf(gv);
        float hn = sigm(ov) * tanhf(cn);
        cbuf[b * H_ + jg]  = cn;
        hnext[b * H_ + jg] = (h16)hn;
    }
}

// MLP kernel: blockIdx.y selects branch: 0 -> p1 (also x_next), 1 -> p2, 2 -> offset.
// Stage 1: t = tanh(h @ W1^T + b1) -> LDS (f16). Stage 2: t @ W2^T + b2 (WMMA / dot).
// h tile staged in LDS once; both GEMM stages read A-fragments from LDS.
__global__ void k_step_mlp(int t,
                           const h16* __restrict__ hcur,
                           const h16* __restrict__ p1W1h, const float* __restrict__ p1b1,
                           const h16* __restrict__ p1W2h, const float* __restrict__ p1b2,
                           const h16* __restrict__ p2W1h, const float* __restrict__ p2b1,
                           const h16* __restrict__ p2W2h, const float* __restrict__ p2b2,
                           const h16* __restrict__ oW1h,  const float* __restrict__ ob1,
                           const h16* __restrict__ oW2h,  const float* __restrict__ ob2,
                           const int* __restrict__ lengths, const int* __restrict__ meta,
                           float* __restrict__ out, h16* __restrict__ xnext) {
    __shared__ alignas(16) h16 h_s[16][LDP_H];
    __shared__ alignas(16) h16 tl[16][LDP_H];
    __shared__ float red[16][17];
    int lane = threadIdx.x & 31, wv = threadIdx.x >> 5;
    int b0 = blockIdx.x * 16;
    int mode = blockIdx.y;

    for (int i = threadIdx.x; i < 16 * 64; i += 256) {
        int row = i >> 6;
        int c8  = (i & 63) << 3;
        *(v8h*)&h_s[row][c8] = *(const v8h*)&hcur[(size_t)(b0 + row) * H_ + c8];
    }
    __syncthreads();

    const h16*   W1 = (mode == 0) ? p1W1h : (mode == 1) ? p2W1h : oW1h;
    const float* b1 = (mode == 0) ? p1b1  : (mode == 1) ? p2b1  : ob1;
    {   // hidden layer: [16 x 512], 8 waves x 64 cols
        int c0 = wv * 64;
        v8f acc[4] = {};
        for (int kc = 0; kc < H_; kc += 32) {
            v16h a = load_frag(&h_s[0][0], LDP_H, 0, kc, lane);
            #pragma unroll
            for (int nt = 0; nt < 4; ++nt) {
                v16h bf = load_frag(W1, H_, c0 + nt * 16, kc, lane);
                acc[nt] = wmma16(a, bf, acc[nt]);
            }
        }
        int rowoff = (lane >> 4) * 8, col_l = lane & 15;
        #pragma unroll
        for (int nt = 0; nt < 4; ++nt) {
            int n = c0 + nt * 16 + col_l;
            float bb = b1[n];
            #pragma unroll
            for (int r = 0; r < 8; ++r)
                tl[rowoff + r][n] = (h16)tanhf(acc[nt][r] + bb);
        }
    }
    __syncthreads();
    int N = meta[TMAX];
    if (mode < 2) {
        const h16*   W2 = mode ? p2W2h : p1W2h;
        const float* b2 = mode ? p2b2  : p1b2;
        float* obase = out + (size_t)mode * N * O_;
        int c0 = wv * 16;
        v8f acc = {};
        for (int kc = 0; kc < H_; kc += 32) {
            v16h a  = load_frag(&tl[0][0], LDP_H, 0, kc, lane);   // A from LDS
            v16h bf = load_frag(W2, H_, c0, kc, lane);
            acc = wmma16(a, bf, acc);
        }
        int rowoff = (lane >> 4) * 8, col_l = lane & 15;
        int n = c0 + col_l;
        float bb = b2[n];
        #pragma unroll
        for (int r = 0; r < 8; ++r) {
            int b = b0 + rowoff + r;
            float v = acc[r] + bb;
            if (mode == 0) xnext[b * O_ + n] = (h16)v;   // next-step input = p1
            if (t < lengths[b])
                obase[(size_t)(meta[t] + b) * O_ + n] = v;
        }
    } else {
        // offset = tl . oW2 + ob2 ; 16 partial lanes per row
        int row = threadIdx.x >> 4, part = threadIdx.x & 15;
        float s = 0.0f;
        for (int j = part * 32; j < part * 32 + 32; ++j)
            s += (float)tl[row][j] * (float)oW2h[j];
        red[row][part] = s;
        __syncthreads();
        if (part == 0) {
            float tot = ob2[0];
            for (int i = 0; i < 16; ++i) tot += red[row][i];
            int b = b0 + row;
            if (t < lengths[b])
                out[(size_t)2 * N * O_ + (meta[t] + b)] = tot;
        }
    }
}

// ---------------- host launch ----------------

extern "C" void kernel_launch(void* const* d_in, const int* in_sizes, int n_in,
                              void* d_out, int out_size, void* d_ws, size_t ws_size,
                              hipStream_t stream) {
    (void)in_sizes; (void)n_in; (void)out_size; (void)ws_size;
    const float* features = (const float*)d_in[0];
    const int*   lengths  = (const int*)d_in[1];
    const float* W_f2h = (const float*)d_in[2];
    const float* b_f2h = (const float*)d_in[3];
    const float* W_ih  = (const float*)d_in[4];
    const float* W_hh  = (const float*)d_in[5];
    const float* b_ih  = (const float*)d_in[6];
    const float* b_hh  = (const float*)d_in[7];
    const float* p1_W1 = (const float*)d_in[8];
    const float* p1_b1 = (const float*)d_in[9];
    const float* p1_W2 = (const float*)d_in[10];
    const float* p1_b2 = (const float*)d_in[11];
    const float* p2_W1 = (const float*)d_in[12];
    const float* p2_b1 = (const float*)d_in[13];
    const float* p2_W2 = (const float*)d_in[14];
    const float* p2_b2 = (const float*)d_in[15];
    const float* oW1   = (const float*)d_in[16];
    const float* ob1   = (const float*)d_in[17];
    const float* oW2   = (const float*)d_in[18];
    const float* ob2   = (const float*)d_in[19];

    char* ws = (char*)d_ws;
    size_t off = 0;
    auto alloc = [&](size_t bytes) -> void* {
        off = (off + 255) & ~(size_t)255;
        void* p = ws + off;
        off += bytes;
        return p;
    };

    h16* feat_h  = (h16*)alloc((size_t)B_ * F_ * 2);
    h16* Wf2h_h  = (h16*)alloc((size_t)2 * H_ * F_ * 2);
    h16* Wih_h   = (h16*)alloc((size_t)4 * H_ * O_ * 2);
    h16* Whh_h   = (h16*)alloc((size_t)4 * H_ * H_ * 2);
    h16* p1W1h   = (h16*)alloc((size_t)M_ * H_ * 2);
    h16* p1W2h   = (h16*)alloc((size_t)O_ * M_ * 2);
    h16* p2W1h   = (h16*)alloc((size_t)M_ * H_ * 2);
    h16* p2W2h   = (h16*)alloc((size_t)O_ * M_ * 2);
    h16* oW1h    = (h16*)alloc((size_t)M_ * H_ * 2);
    h16* oW2h    = (h16*)alloc((size_t)M_ * 2);
    h16* hbuf0   = (h16*)alloc((size_t)B_ * H_ * 2);
    h16* hbuf1   = (h16*)alloc((size_t)B_ * H_ * 2);
    float* cbuf  = (float*)alloc((size_t)B_ * H_ * 4);
    h16* xbuf    = (h16*)alloc((size_t)B_ * O_ * 2);
    int* meta    = (int*)alloc((size_t)(TMAX + 1) * 4);

    auto conv = [&](const float* s, h16* d, int n) {
        k_f32_to_f16<<<(n + 255) / 256, 256, 0, stream>>>(s, d, n);
    };
    conv(features, feat_h, B_ * F_);
    conv(W_f2h, Wf2h_h, 2 * H_ * F_);
    conv(W_ih,  Wih_h,  4 * H_ * O_);
    conv(W_hh,  Whh_h,  4 * H_ * H_);
    conv(p1_W1, p1W1h, M_ * H_);
    conv(p1_W2, p1W2h, O_ * M_);
    conv(p2_W1, p2W1h, M_ * H_);
    conv(p2_W2, p2W2h, O_ * M_);
    conv(oW1,   oW1h,  M_ * H_);
    conv(oW2,   oW2h,  M_);

    k_meta<<<1, TMAX, 0, stream>>>(lengths, meta);
    k_zero_x<<<(B_ * O_ + 255) / 256, 256, 0, stream>>>(xbuf);
    k_init<<<dim3(B_ / 16, 2), 256, 0, stream>>>(feat_h, Wf2h_h, b_f2h, hbuf0, cbuf);

    float* outf = (float*)d_out;
    for (int t = 0; t < TMAX; ++t) {
        const h16* hp = (t & 1) ? hbuf1 : hbuf0;
        h16*       hn = (t & 1) ? hbuf0 : hbuf1;
        k_step_gates<<<dim3(B_ / 16, H_ / 64), 256, 0, stream>>>(
            xbuf, hp, hn, cbuf, Wih_h, Whh_h, b_ih, b_hh);
        k_step_mlp<<<dim3(B_ / 16, 3), 256, 0, stream>>>(
            t, hn, p1W1h, p1_b1, p1W2h, p1_b2, p2W1h, p2_b1, p2W2h, p2_b2,
            oW1h, ob1, oW2h, ob2, lengths, meta, outf, xbuf);
    }
}
